// Conv2d_33767032881634
// MI455X (gfx1250) — matrix operand order repaired and checked by reading, not compiled, not run on hardware
//
#include <hip/hip_runtime.h>

// ---------------- types for WMMA ----------------
typedef __bf16 bf16_t;
typedef bf16_t v16bf __attribute__((ext_vector_type(16)));
typedef bf16_t v8bf  __attribute__((ext_vector_type(8)));
typedef float  v8f   __attribute__((ext_vector_type(8)));

// ---------------- problem constants ----------------
#define CI      128
#define CO      256
#define HH      56
#define WW      56
#define HWPIX   3136        // 56*56
#define KDIM    1152        // CI*3*3
#define MT      64          // C_out tile per block
#define NT      128         // pixel tile per block
#define KT      32          // K step (bf16 WMMA K)
#define KSTRIDE 40          // KT + 8 pad -> 80B row stride (16B aligned, bank-friendly)

#define XPLANE_ELEMS ((size_t)32 * CI * HWPIX)    // 12,845,056 ushorts per plane
#define WPLANE_ELEMS ((size_t)CO * KDIM)          // 294,912 ushorts per plane
#define WPLANE_BYTES (WPLANE_ELEMS * 2)           // 589,824  (16B multiple)
#define XPLANE_BYTES (XPLANE_ELEMS * 2)           // 25,690,112 (16B multiple)
#define WS_NEEDED    (2 * WPLANE_BYTES + 2 * XPLANE_BYTES)

// bf16 RNE split helpers (bit level)
__device__ __forceinline__ unsigned short f2bf(float f) {
    unsigned int u = __float_as_uint(f);
    unsigned int lsb = (u >> 16) & 1u;
    u += 0x7fffu + lsb;
    return (unsigned short)(u >> 16);
}
__device__ __forceinline__ float bf2f(unsigned short s) {
    return __uint_as_float(((unsigned int)s) << 16);
}
// pack f32 -> (bf16_hi) | (bf16_lo << 16)
__device__ __forceinline__ unsigned int packsplit(float f) {
    unsigned short hi = f2bf(f);
    unsigned short lo = f2bf(f - bf2f(hi));
    return (unsigned int)hi | ((unsigned int)lo << 16);
}

// ================= prep kernels =================
// x: NCHW fp32 -> two NHWC bf16 planes (hi, lo) via 32x32 LDS tile transpose.
//   xhi[img][p][ci], xlo[img][p][ci]
__global__ __launch_bounds__(256)
void prep_x_nhwc(const float* __restrict__ x,
                 unsigned short* __restrict__ xhi,
                 unsigned short* __restrict__ xlo) {
    __shared__ unsigned int sT[32][33];           // packed (hi|lo), padded transpose tile
    const int img = blockIdx.z;
    const int gp0 = blockIdx.x * 32;              // pixel tile  (98 tiles)
    const int gc0 = blockIdx.y * 32;              // channel tile (4 tiles)

    {
        const int col = threadIdx.x & 31;
        const int row = threadIdx.x >> 5;         // 0..7
        const float* xi = x + ((size_t)img * CI + gc0) * HWPIX + gp0;
#pragma unroll
        for (int i = 0; i < 4; ++i) {
            int ci_l = row + i * 8;
            sT[ci_l][col] = packsplit(xi[(size_t)ci_l * HWPIX + col]);  // coalesced read
        }
    }
    __syncthreads();
    {
        const int cpair = threadIdx.x & 15;       // channel pair 0..15 (covers 32 ci)
        const int prow  = threadIdx.x >> 4;       // 0..15
        size_t base = ((size_t)img * HWPIX + gp0) * CI + gc0 + cpair * 2;
#pragma unroll
        for (int i = 0; i < 2; ++i) {
            int p_l = prow + i * 16;
            unsigned int v0 = sT[cpair * 2][p_l];
            unsigned int v1 = sT[cpair * 2 + 1][p_l];
            unsigned int hi = (v0 & 0xffffu) | (v1 << 16);
            unsigned int lo = (v0 >> 16) | (v1 & 0xffff0000u);
            size_t o = base + (size_t)p_l * CI;
            *(unsigned int*)(xhi + o) = hi;       // coalesced 4B stores
            *(unsigned int*)(xlo + o) = lo;
        }
    }
}

// weights: split + permute K from (ci,kh,kw) to (kh,kw)-major, two planes:
//   whi/wlo[co][(kh*3+kw)*128 + ci]
__global__ __launch_bounds__(256)
void prep_w(const float* __restrict__ wgt,
            unsigned short* __restrict__ whi,
            unsigned short* __restrict__ wlo) {
    int idx = blockIdx.x * blockDim.x + threadIdx.x;   // over CO*KDIM
    if (idx >= (int)WPLANE_ELEMS) return;
    int co = idx / KDIM;
    int t  = idx - co * KDIM;
    int r  = t >> 7;          // kh*3+kw  (0..8)
    int ci = t & 127;
    float f = wgt[(size_t)co * KDIM + ci * 9 + r];
    unsigned short hi = f2bf(f);
    whi[idx] = hi;
    wlo[idx] = f2bf(f - bf2f(hi));
}

// ================= fragment loaders =================
// A 16x32 bf16: lane = row (lane%16); elems 0..7 -> K=half*8+0..7, 8..15 -> K=16+half*8+0..7
__device__ __forceinline__ v16bf frag_a(const unsigned short* rowp, int half) {
    const v8bf* p0 = (const v8bf*)(rowp + half * 8);
    const v8bf* p1 = (const v8bf*)(rowp + 16 + half * 8);
    v8bf a = *p0, b = *p1;
    return __builtin_shufflevector(a, b, 0,1,2,3,4,5,6,7,8,9,10,11,12,13,14,15);
}
// B 32x16 bf16: lane = col (lane%16); elems 0..7 -> K=half*8+0..7, 8..15 -> K=16+half*8+0..7
__device__ __forceinline__ v16bf frag_b(const unsigned short* rowp, int half) {
    const v8bf* p0 = (const v8bf*)(rowp + half * 8);
    const v8bf* p1 = (const v8bf*)(rowp + 16 + half * 8);
    v8bf a = *p0, b = *p1;
    return __builtin_shufflevector(a, b, 0,1,2,3,4,5,6,7,8,9,10,11,12,13,14,15);
}

// ================= main kernel (pre-split NHWC planes; hot loop = pure copy) ===
__global__ __launch_bounds__(256)
void conv3x3_wmma_main(const unsigned short* __restrict__ xhi,
                       const unsigned short* __restrict__ xlo,
                       const unsigned short* __restrict__ whi,
                       const unsigned short* __restrict__ wlo,
                       const float* __restrict__ bias,
                       float* __restrict__ out)
{
    __shared__ unsigned short sAhi[MT * KSTRIDE];
    __shared__ unsigned short sAlo[MT * KSTRIDE];
    __shared__ unsigned short sBhi[NT * KSTRIDE];
    __shared__ unsigned short sBlo[NT * KSTRIDE];

    const int tid      = threadIdx.x;
    const int pix_base = blockIdx.x * NT;
    const int co_base  = blockIdx.y * MT;
    const int img      = blockIdx.z;

    const int wid  = tid >> 5;
    const int lane = tid & 31;
    const int half = lane >> 4;
    const int lm   = lane & 15;
    const int wm   = wid & 1;                 // wave grid 2 (M) x 4 (N)
    const int wn   = wid >> 1;

    v8f acc00 = 0.0f, acc01 = 0.0f, acc10 = 0.0f, acc11 = 0.0f;  // [sm][sn]

    const size_t ximg_off = (size_t)img * HWPIX * CI;

    // A-tile loader: 64 rows x 4 chunks of 8
    const int a_row = tid >> 2;
    const int a_col = (tid & 3) * 8;
    const size_t a_goff = (size_t)(co_base + a_row) * KDIM + a_col;
    unsigned short* const a_dh = sAhi + a_row * KSTRIDE + a_col;
    unsigned short* const a_dl = sAlo + a_row * KSTRIDE + a_col;

    // B-tile loader: 128 columns (pixels) x 2 groups of 16 k(ci)-rows
    const int b_col  = tid & 127;
    const int b_krow = (tid >> 7) * 16;
    const int p      = pix_base + b_col;
    const int ph     = p / WW;
    const int pw     = p - ph * WW;
    unsigned short* const b_dh = sBhi + b_col * KSTRIDE + b_krow;
    unsigned short* const b_dl = sBlo + b_col * KSTRIDE + b_krow;

    // per-thread 9-bit halo validity mask (r = kh*3+kw)
    unsigned int mask9 = 0;
    if (p < HWPIX) {
#pragma unroll
        for (int r = 0; r < 9; ++r) {
            int kh = r / 3, kw = r - (r / 3) * 3;
            int ih = ph + kh - 1, iw = pw + kw - 1;
            if ((unsigned)ih < (unsigned)HH && (unsigned)iw < (unsigned)WW)
                mask9 |= (1u << r);
        }
    }

    const int rowA0 = (wm * 32 + lm) * KSTRIDE;
    const int rowA1 = (wm * 32 + 16 + lm) * KSTRIDE;

    for (int kk = 0; kk < KDIM; kk += KT) {
        // ---- stage A: straight copy (global b128 -> LDS b128) ----
        {
            uint4 vh = *(const uint4*)(whi + a_goff + kk);
            uint4 vl = *(const uint4*)(wlo + a_goff + kk);
            *(uint4*)a_dh = vh;
            *(uint4*)a_dl = vl;
        }
        // ---- stage B: straight copy, uniform (kh,kw) halo per k-step ----
        {
            int r   = kk >> 7;                    // kh*3+kw (uniform)
            int kh  = r / 3;
            int kw  = r - kh * 3;
            int off = (kh - 1) * WW + (kw - 1);
            int cib = kk & 127;
            bool valid = (mask9 >> r) & 1u;
            uint4 h0, h1, l0, l1;
            if (valid) {
                size_t o = ximg_off + (size_t)(p + off) * CI + cib + b_krow;
                h0 = *(const uint4*)(xhi + o);
                h1 = *(const uint4*)(xhi + o + 8);
                l0 = *(const uint4*)(xlo + o);
                l1 = *(const uint4*)(xlo + o + 8);
            } else {
                h0 = h1 = l0 = l1 = make_uint4(0u, 0u, 0u, 0u);
            }
            *(uint4*)(b_dh)     = h0;
            *(uint4*)(b_dh + 8) = h1;
            *(uint4*)(b_dl)     = l0;
            *(uint4*)(b_dl + 8) = l1;
        }
        __syncthreads();

        // ---- A fragments resident; B fragments streamed per column ----
        v16bf aHi0 = frag_a(sAhi + rowA0, half);
        v16bf aHi1 = frag_a(sAhi + rowA1, half);
        v16bf aLo0 = frag_a(sAlo + rowA0, half);
        v16bf aLo1 = frag_a(sAlo + rowA1, half);

#pragma unroll
        for (int sn = 0; sn < 2; ++sn) {
            int rowB = (wn * 32 + sn * 16 + lm) * KSTRIDE;
            v16bf bh = frag_b(sBhi + rowB, half);
            v16bf bl = frag_b(sBlo + rowB, half);
            v8f& c0 = sn ? acc01 : acc00;
            v8f& c1 = sn ? acc11 : acc10;
            // interleave sm to space D-register RAW hazards
            c0 = __builtin_amdgcn_wmma_f32_16x16x32_bf16(false, aHi0, false, bl, (short)0, c0, false, false);
            c1 = __builtin_amdgcn_wmma_f32_16x16x32_bf16(false, aHi1, false, bl, (short)0, c1, false, false);
            c0 = __builtin_amdgcn_wmma_f32_16x16x32_bf16(false, aLo0, false, bh, (short)0, c0, false, false);
            c1 = __builtin_amdgcn_wmma_f32_16x16x32_bf16(false, aLo1, false, bh, (short)0, c1, false, false);
            c0 = __builtin_amdgcn_wmma_f32_16x16x32_bf16(false, aHi0, false, bh, (short)0, c0, false, false);
            c1 = __builtin_amdgcn_wmma_f32_16x16x32_bf16(false, aHi1, false, bh, (short)0, c1, false, false);
        }
        __syncthreads();
    }

    // ---- epilogue ----
#pragma unroll
    for (int sm = 0; sm < 2; ++sm) {
#pragma unroll
        for (int sn = 0; sn < 2; ++sn) {
            const v8f& a = sm ? (sn ? acc11 : acc10) : (sn ? acc01 : acc00);
            int pcol = pix_base + wn * 32 + sn * 16 + lm;
            if (pcol >= HWPIX) continue;
            int crow0 = co_base + wm * 32 + sm * 16 + half * 8;
#pragma unroll
            for (int i = 0; i < 8; ++i) {
                int c = crow0 + i;
                out[((size_t)img * CO + c) * HWPIX + pcol] = a[i] + bias[c];
            }
        }
    }
}

// ================= fallback kernel (self-converting, no workspace) =================
__global__ __launch_bounds__(256)
void conv3x3_wmma_fallback(const float* __restrict__ x,
                           const float* __restrict__ wgt,
                           const float* __restrict__ bias,
                           float* __restrict__ out)
{
    __shared__ unsigned short sAhi[MT * KSTRIDE];
    __shared__ unsigned short sAlo[MT * KSTRIDE];
    __shared__ unsigned short sBhi[NT * KSTRIDE];
    __shared__ unsigned short sBlo[NT * KSTRIDE];

    const int tid      = threadIdx.x;
    const int pix_base = blockIdx.x * NT;
    const int co_base  = blockIdx.y * MT;
    const int img      = blockIdx.z;

    const int wid  = tid >> 5;
    const int lane = tid & 31;
    const int half = lane >> 4;
    const int lm   = lane & 15;
    const int wm   = wid & 1;
    const int wn   = wid >> 1;

    v8f acc00 = 0.0f, acc01 = 0.0f, acc10 = 0.0f, acc11 = 0.0f;

    const float* ximg = x + (size_t)img * CI * HWPIX;

    const int a_row = tid >> 2;
    const int a_col = (tid & 3) * 8;
    const int b_col  = tid & 127;
    const int b_krow = (tid >> 7) * 16;
    const int p      = pix_base + b_col;
    const int ph     = p / WW;
    const int pw     = p - ph * WW;

    const int rowA0 = (wm * 32 + lm) * KSTRIDE;
    const int rowA1 = (wm * 32 + 16 + lm) * KSTRIDE;

    for (int kk = 0; kk < KDIM; kk += KT) {
        {
            const float* wp = wgt + (size_t)(co_base + a_row) * KDIM + kk + a_col;
            unsigned short* dh = sAhi + a_row * KSTRIDE + a_col;
            unsigned short* dl = sAlo + a_row * KSTRIDE + a_col;
#pragma unroll
            for (int j = 0; j < 8; ++j) {
                float f = wp[j];
                unsigned short hi = f2bf(f);
                dh[j] = hi;
                dl[j] = f2bf(f - bf2f(hi));
            }
        }
        {
            unsigned short* dh = sBhi + b_col * KSTRIDE + b_krow;
            unsigned short* dl = sBlo + b_col * KSTRIDE + b_krow;
#pragma unroll
            for (int j = 0; j < 16; ++j) {
                int k  = kk + b_krow + j;
                int ci = k / 9;
                int rr = k - ci * 9;
                int kh = rr / 3;
                int kw = rr - kh * 3;
                float f = 0.0f;
                if (p < HWPIX) {
                    int ih = ph + kh - 1;
                    int iw = pw + kw - 1;
                    if ((unsigned)ih < (unsigned)HH && (unsigned)iw < (unsigned)WW)
                        f = ximg[(ci * HH + ih) * WW + iw];
                }
                unsigned short hi = f2bf(f);
                dh[j] = hi;
                dl[j] = f2bf(f - bf2f(hi));
            }
        }
        __syncthreads();

        v16bf aHi0 = frag_a(sAhi + rowA0, half);
        v16bf aHi1 = frag_a(sAhi + rowA1, half);
        v16bf aLo0 = frag_a(sAlo + rowA0, half);
        v16bf aLo1 = frag_a(sAlo + rowA1, half);

#pragma unroll
        for (int sn = 0; sn < 2; ++sn) {
            int rowB = (wn * 32 + sn * 16 + lm) * KSTRIDE;
            v16bf bh = frag_b(sBhi + rowB, half);
            v16bf bl = frag_b(sBlo + rowB, half);
            v8f& c0 = sn ? acc01 : acc00;
            v8f& c1 = sn ? acc11 : acc10;
            c0 = __builtin_amdgcn_wmma_f32_16x16x32_bf16(false, aHi0, false, bl, (short)0, c0, false, false);
            c1 = __builtin_amdgcn_wmma_f32_16x16x32_bf16(false, aHi1, false, bl, (short)0, c1, false, false);
            c0 = __builtin_amdgcn_wmma_f32_16x16x32_bf16(false, aLo0, false, bh, (short)0, c0, false, false);
            c1 = __builtin_amdgcn_wmma_f32_16x16x32_bf16(false, aLo1, false, bh, (short)0, c1, false, false);
            c0 = __builtin_amdgcn_wmma_f32_16x16x32_bf16(false, aHi0, false, bh, (short)0, c0, false, false);
            c1 = __builtin_amdgcn_wmma_f32_16x16x32_bf16(false, aHi1, false, bh, (short)0, c1, false, false);
        }
        __syncthreads();
    }

#pragma unroll
    for (int sm = 0; sm < 2; ++sm) {
#pragma unroll
        for (int sn = 0; sn < 2; ++sn) {
            const v8f& a = sm ? (sn ? acc11 : acc10) : (sn ? acc01 : acc00);
            int pcol = pix_base + wn * 32 + sn * 16 + lm;
            if (pcol >= HWPIX) continue;
            int crow0 = co_base + wm * 32 + sm * 16 + half * 8;
#pragma unroll
            for (int i = 0; i < 8; ++i) {
                int c = crow0 + i;
                out[((size_t)img * CO + c) * HWPIX + pcol] = a[i] + bias[c];
            }
        }
    }
}

extern "C" void kernel_launch(void* const* d_in, const int* in_sizes, int n_in,
                              void* d_out, int out_size, void* d_ws, size_t ws_size,
                              hipStream_t stream) {
    (void)in_sizes; (void)n_in; (void)out_size;
    const float* x    = (const float*)d_in[0];
    const float* wgt  = (const float*)d_in[1];
    const float* bias = (const float*)d_in[2];
    float* out        = (float*)d_out;

    dim3 grid((HWPIX + NT - 1) / NT, CO / MT, 32);
    dim3 block(256);

    if (d_ws != nullptr && ws_size >= WS_NEEDED) {
        char* wsp = (char*)d_ws;
        unsigned short* whi = (unsigned short*)(wsp);
        unsigned short* wlo = (unsigned short*)(wsp + WPLANE_BYTES);
        unsigned short* xhi = (unsigned short*)(wsp + 2 * WPLANE_BYTES);
        unsigned short* xlo = (unsigned short*)(wsp + 2 * WPLANE_BYTES + XPLANE_BYTES);

        int wblocks = (int)((WPLANE_ELEMS + 255) / 256);
        prep_w<<<wblocks, 256, 0, stream>>>(wgt, whi, wlo);

        dim3 xgrid(HWPIX / 32, CI / 32, 32);      // 98 x 4 x 32 tiles
        prep_x_nhwc<<<xgrid, 256, 0, stream>>>(x, xhi, xlo);

        conv3x3_wmma_main<<<grid, block, 0, stream>>>(xhi, xlo, whi, wlo, bias, out);
    } else {
        conv3x3_wmma_fallback<<<grid, block, 0, stream>>>(x, wgt, bias, out);
    }
}
